// MoEProjector_9852654977535
// MI455X (gfx1250) — matrix operand order repaired and checked by reading, not compiled, not run on hardware
//
#include <hip/hip_runtime.h>
#include <hip/hip_bf16.h>

#define B_TOK 8192
#define DDIM  2048
#define ODIM  2048
#define NEXP  8

typedef __bf16 bf16_t;
typedef __bf16 v16bf __attribute__((ext_vector_type(16)));
typedef float  v8f   __attribute__((ext_vector_type(8)));

union FragBF { uint4 u[2]; v16bf v; };

// ---------------------------------------------------------------------------
// Kernel 1: router. One wave (32 lanes) per token: 8 dot products of length D,
// wave32 shuffle reduction, top-2 + softmax on lane 0, build gather lists.
// ---------------------------------------------------------------------------
__global__ __launch_bounds__(256)
void moe_router_topk(const float* __restrict__ x,
                     const float* __restrict__ rw,
                     const float* __restrict__ rb,
                     int*   __restrict__ counts,   // [2][E]
                     int*   __restrict__ lists,    // [2][E][B]
                     float* __restrict__ wlists)   // [2][E][B]
{
    const int lane = threadIdx.x & 31;
    const int wid  = threadIdx.x >> 5;
    const int tok  = blockIdx.x * 8 + wid;
    if (tok >= B_TOK) return;

    const float* xr = x + (size_t)tok * DDIM;
    float acc[NEXP];
#pragma unroll
    for (int e = 0; e < NEXP; ++e) acc[e] = 0.f;

    for (int d = lane; d < DDIM; d += 32) {
        float xv = xr[d];
#pragma unroll
        for (int e = 0; e < NEXP; ++e)
            acc[e] = fmaf(xv, rw[(size_t)e * DDIM + d], acc[e]);
    }
#pragma unroll
    for (int e = 0; e < NEXP; ++e) {
#pragma unroll
        for (int off = 16; off > 0; off >>= 1)
            acc[e] += __shfl_xor(acc[e], off, 32);
    }

    if (lane == 0) {
        float l[NEXP];
#pragma unroll
        for (int e = 0; e < NEXP; ++e) l[e] = acc[e] + rb[e];

        // top-1 then top-2, first index wins ties (matches jax top_k)
        int e0 = 0;
#pragma unroll
        for (int e = 1; e < NEXP; ++e) if (l[e] > l[e0]) e0 = e;
        int e1 = (e0 == 0) ? 1 : 0;
#pragma unroll
        for (int e = 0; e < NEXP; ++e)
            if (e != e0 && l[e] > l[e1]) e1 = e;

        // softmax over {l[e0], l[e1]} with l[e0] as max
        float z1 = __expf(l[e1] - l[e0]);
        float s  = 1.0f + z1;
        float w0 = 1.0f / s;
        float w1 = z1 / s;

        int p0 = atomicAdd(&counts[0 * NEXP + e0], 1);
        lists [(size_t)(0 * NEXP + e0) * B_TOK + p0] = tok;
        wlists[(size_t)(0 * NEXP + e0) * B_TOK + p0] = w0;

        int p1 = atomicAdd(&counts[1 * NEXP + e1], 1);
        lists [(size_t)(1 * NEXP + e1) * B_TOK + p1] = tok;
        wlists[(size_t)(1 * NEXP + e1) * B_TOK + p1] = w1;
    }
}

// ---------------------------------------------------------------------------
// Helper: convert 16 fp32 -> 16 bf16 and store as two 16B LDS writes.
// ---------------------------------------------------------------------------
__device__ inline void cvt_store16(bf16_t* dst, float4 a, float4 b, float4 c, float4 d)
{
    union { bf16_t h[16]; uint4 u[2]; } t;
    t.h[0]  = (bf16_t)a.x; t.h[1]  = (bf16_t)a.y; t.h[2]  = (bf16_t)a.z; t.h[3]  = (bf16_t)a.w;
    t.h[4]  = (bf16_t)b.x; t.h[5]  = (bf16_t)b.y; t.h[6]  = (bf16_t)b.z; t.h[7]  = (bf16_t)b.w;
    t.h[8]  = (bf16_t)c.x; t.h[9]  = (bf16_t)c.y; t.h[10] = (bf16_t)c.z; t.h[11] = (bf16_t)c.w;
    t.h[12] = (bf16_t)d.x; t.h[13] = (bf16_t)d.y; t.h[14] = (bf16_t)d.z; t.h[15] = (bf16_t)d.w;
    uint4* du = (uint4*)dst;
    du[0] = t.u[0];
    du[1] = t.u[1];
}

// ---------------------------------------------------------------------------
// Kernel 2/3: gathered per-expert GEMM. 128x128 output tile per block,
// 8 waves, each wave = 32x64 (2 M-tiles x 4 N-tiles of 16x16),
// K-loop over D in steps of 32 with v_wmma_f32_16x16x32_bf16.
// slot 0 writes, slot 1 accumulates (slot-disjoint rows -> no collisions).
// ---------------------------------------------------------------------------
__global__ __launch_bounds__(256)
void moe_expert_gemm(const float* __restrict__ x,
                     const float* __restrict__ ew,   // [E][O][D]
                     const float* __restrict__ eb,   // [E][O]
                     float*       __restrict__ out,  // [B][O]
                     const int*   __restrict__ counts,
                     const int*   __restrict__ lists,
                     const float* __restrict__ wlists,
                     int slot, int accum)
{
    __shared__ bf16_t sA[128][40];   // 128 rows x 32 K, padded to 40 (80B stride)
    __shared__ bf16_t sB[128][40];
    __shared__ int    sIdx[128];
    __shared__ float  sW[128];

    const int e    = blockIdx.z;
    const int cnt  = counts[slot * NEXP + e];
    const int row0 = blockIdx.x * 128;
    if (row0 >= cnt) return;
    const int n0   = blockIdx.y * 128;
    const int t    = threadIdx.x;
    const int rows_valid = min(128, cnt - row0);

    const int*   lst = lists  + (size_t)(slot * NEXP + e) * B_TOK;
    const float* wl  = wlists + (size_t)(slot * NEXP + e) * B_TOK;

    if (t < 128) {
        int g = row0 + t;
        if (g < cnt) { sIdx[t] = lst[g]; sW[t] = wl[g]; }
        else         { sIdx[t] = 0;      sW[t] = 0.f;  }
    }
    __syncthreads();

    // global->LDS staging assignment: 2 threads per row, 16 floats each
    const int    lr   = t >> 1;            // 0..127
    const int    kq   = (t & 1) * 16;      // 0 or 16
    const size_t arow = (size_t)sIdx[lr] * DDIM;
    const size_t brow = ((size_t)e * ODIM + (n0 + lr)) * (size_t)DDIM;

    // wave tiling
    const int lane  = t & 31;
    const int wid   = t >> 5;
    const int mbase = (wid >> 1) * 32;     // 0,32,64,96
    const int nbase = (wid & 1) * 64;      // 0,64
    const int hh    = lane >> 4;           // half-wave select
    const int ln    = lane & 15;

    v8f acc[2][4];
#pragma unroll
    for (int tm = 0; tm < 2; ++tm)
#pragma unroll
        for (int tn = 0; tn < 4; ++tn)
            acc[tm][tn] = {};

    for (int k0 = 0; k0 < DDIM; k0 += 32) {
        const float4* pa = (const float4*)(x  + arow + k0 + kq);
        const float4* pb = (const float4*)(ew + brow + k0 + kq);
        float4 a0 = pa[0], a1 = pa[1], a2 = pa[2], a3 = pa[3];
        float4 b0 = pb[0], b1 = pb[1], b2 = pb[2], b3 = pb[3];
        if (k0 + 32 < DDIM) {
            __builtin_prefetch(x  + arow + k0 + 32 + kq, 0, 3);
            __builtin_prefetch(ew + brow + k0 + 32 + kq, 0, 3);
        }
        cvt_store16(&sA[lr][kq], a0, a1, a2, a3);
        cvt_store16(&sB[lr][kq], b0, b1, b2, b3);
        __syncthreads();

        // A frags: lane (hh,ln): row = m, elements K = {8*hh..8*hh+7, 16+8*hh..23+8*hh}
        FragBF fa[2];
#pragma unroll
        for (int tm = 0; tm < 2; ++tm) {
            const bf16_t* rp = &sA[mbase + tm * 16 + ln][0];
            fa[tm].u[0] = *(const uint4*)(rp + 8 * hh);
            fa[tm].u[1] = *(const uint4*)(rp + 16 + 8 * hh);
        }
        // B frags: lane (hh,ln): col N = ln (row of W tile), K = {8*hh..8*hh+7, 16+8*hh..23+8*hh}
        FragBF fb[4];
#pragma unroll
        for (int tn = 0; tn < 4; ++tn) {
            const bf16_t* rp = &sB[nbase + tn * 16 + ln][8 * hh];
            fb[tn].u[0] = *(const uint4*)(rp);
            fb[tn].u[1] = *(const uint4*)(rp + 16);
        }
#pragma unroll
        for (int tm = 0; tm < 2; ++tm)
#pragma unroll
            for (int tn = 0; tn < 4; ++tn)
                acc[tm][tn] = __builtin_amdgcn_wmma_f32_16x16x32_bf16(
                    false, fa[tm].v, false, fb[tn].v,
                    (short)0, acc[tm][tn], false, false);
        __syncthreads();
    }

    // Epilogue: D layout lane (hh,ln): col = ln, VGPR i -> row = 8*hh + i
#pragma unroll
    for (int tm = 0; tm < 2; ++tm) {
#pragma unroll
        for (int tn = 0; tn < 4; ++tn) {
            const int col = n0 + nbase + tn * 16 + ln;
            const float bv = eb[(size_t)e * ODIM + col];
            v8f c = acc[tm][tn];
#pragma unroll
            for (int i = 0; i < 8; ++i) {
                int r_tile = mbase + tm * 16 + hh * 8 + i;
                if (r_tile < rows_valid) {
                    int   tokid = sIdx[r_tile];
                    float wv    = sW[r_tile];
                    float val   = wv * (c[i] + bv);
                    float* po = &out[(size_t)tokid * ODIM + col];
                    if (accum) *po += val;
                    else       *po  = val;
                }
            }
        }
    }
}

extern "C" void kernel_launch(void* const* d_in, const int* in_sizes, int n_in,
                              void* d_out, int out_size, void* d_ws, size_t ws_size,
                              hipStream_t stream)
{
    const float* x  = (const float*)d_in[0];
    const float* rw = (const float*)d_in[1];
    const float* rb = (const float*)d_in[2];
    const float* ew = (const float*)d_in[3];
    const float* eb = (const float*)d_in[4];
    float* out = (float*)d_out;

    // workspace layout: counts[16] | lists[2*E*B] ints | wlists[2*E*B] floats
    int*   counts = (int*)d_ws;
    int*   lists  = (int*)((char*)d_ws + 64);
    float* wlists = (float*)((char*)d_ws + 64 + (size_t)2 * NEXP * B_TOK * sizeof(int));

    hipMemsetAsync(counts, 0, 64, stream);

    moe_router_topk<<<B_TOK / 8, 256, 0, stream>>>(x, rw, rb, counts, lists, wlists);

    dim3 grid(B_TOK / 128, ODIM / 128, NEXP);
    moe_expert_gemm<<<grid, 256, 0, stream>>>(x, ew, eb, out, counts, lists, wlists, 0, 0);
    moe_expert_gemm<<<grid, 256, 0, stream>>>(x, ew, eb, out, counts, lists, wlists, 1, 1);
}
